// DynamicABPINN_12309376270576
// MI455X (gfx1250) — matrix operand order repaired and checked by reading, not compiled, not run on hardware
//
#include <hip/hip_runtime.h>
#include <hip/hip_bf16.h>

typedef _Float16 half_t;
typedef __attribute__((ext_vector_type(16))) _Float16 v16h;
typedef __attribute__((ext_vector_type(8)))  _Float16 v8h;
typedef __attribute__((ext_vector_type(4)))  _Float16 v4h;
typedef __attribute__((ext_vector_type(8)))  float    v8f;
typedef __attribute__((ext_vector_type(4)))  float    v4f;

#define PI_F 3.14159265358979323846f

union VA { v16h v; v8h h[2]; };

// Branchless tanh: prefer gfx1250 V_TANH_F32; fallback is 5 trans/valu ops,
// saturates correctly for large |x| (exp2 -> inf -> rcp -> 0 -> result 1).
__device__ __forceinline__ float fast_tanh(float v) {
#if __has_builtin(__builtin_amdgcn_tanhf)
    return __builtin_amdgcn_tanhf(v);
#else
    float ax = __builtin_fabsf(v);
    float e  = __builtin_amdgcn_exp2f(ax * 2.8853900817779268f); // 2*log2(e)
    float r  = 1.0f - 2.0f * __builtin_amdgcn_rcpf(e + 1.0f);
    return __builtin_copysignf(r, v);
#endif
}

// One block = 8 waves = 256 points. Subnet loop at block scope stages that
// subnet's hidden weights (f32->f16) into LDS; each wave runs a 32-point tile
// through L2..L4 with v_wmma_f32_16x16x32_f16 (zero C, f32 accum, bias folded
// into the activation FMA), L1/L5 in VALU.
__global__ __launch_bounds__(256) void pinn_kernel(
    const float* __restrict__ x,  const float* __restrict__ t,
    const float* __restrict__ W1, const float* __restrict__ b1,
    const float* __restrict__ W2, const float* __restrict__ b2,
    const float* __restrict__ W3, const float* __restrict__ b3,
    const float* __restrict__ W4, const float* __restrict__ b4,
    const float* __restrict__ W5, const float* __restrict__ b5,
    const float* __restrict__ scales, const float* __restrict__ centers,
    const float* __restrict__ lgam, float* __restrict__ out, int n)
{
    __shared__ __align__(16) half_t s_w[3 * 64 * 64];   // W2,W3,W4 (f16)  24KB
    __shared__ __align__(16) half_t s_act[8][32 * 64];  // per-wave tiles  32KB
    __shared__ __align__(16) float  s_w1p[64 * 4];      // (w0,w1,w2,b1) per neuron
    __shared__ __align__(16) float  s_b[3 * 64];        // b2..b4
    __shared__ __align__(16) float  s_w5[64];
    __shared__ float s_misc[8];                         // b5, scales[0..3]

    const int tid  = threadIdx.x;
    const int lane = tid & 31;
    const int wv   = tid >> 5;
    const int r    = lane & 15;
    const int hi   = lane >> 4;      // 0: lanes 0-15, 1: lanes 16-31

    const int p  = blockIdx.x * 256 + wv * 32 + lane;
    const int pc = (p < n) ? p : (n - 1);        // keep all lanes live (WMMA needs EXEC all-1s)
    const float xv = x[pc];
    const float tv = t[pc];
    const float f0 = __cosf(PI_F * xv);          // periodic embedding (branchless HW trig)
    const float f1 = __sinf(PI_F * xv);

    // Softmax gating: precompute max logit and denominator (logits recomputed
    // per subnet later to avoid a register-indexed private array).
    float lmax = -1e30f;
    #pragma unroll
    for (int k = 0; k < 8; ++k) {
        float gx = __expf(lgam[2 * k]);
        float gt = __expf(lgam[2 * k + 1]);
        float dx = xv - centers[2 * k];
        float dt = tv - centers[2 * k + 1];
        float lk = -(gx * dx * dx + gt * dt * dt);
        lmax = fmaxf(lmax, lk);
    }
    float S = 0.f;
    #pragma unroll
    for (int k = 0; k < 8; ++k) {
        float gx = __expf(lgam[2 * k]);
        float gt = __expf(lgam[2 * k + 1]);
        float dx = xv - centers[2 * k];
        float dt = tv - centers[2 * k + 1];
        S += __expf(-(gx * dx * dx + gt * dt * dt) - lmax);
    }
    const float invS = 1.0f / S;

    float u = 0.f;
    half_t* act = s_act[wv];
    const v8f vzero = {};            // zero C for WMMA chains (hoisted once)

    #pragma unroll 1
    for (int k = 0; k < 8; ++k) {
        __syncthreads();   // previous subnet's compute finished before restaging
        // ---- stage subnet-k weights into LDS (vectorized f32 -> f16) ----
        #pragma unroll
        for (int j = 0; j < 4; ++j) {
            const int i = j * 1024 + tid * 4;    // lanes stride 16B -> coalesced b128
            v4f g2 = *(const v4f*)(W2 + k * 4096 + i);
            v4f g3 = *(const v4f*)(W3 + k * 4096 + i);
            v4f g4 = *(const v4f*)(W4 + k * 4096 + i);
            v4h h2 = {(half_t)g2[0], (half_t)g2[1], (half_t)g2[2], (half_t)g2[3]};
            v4h h3 = {(half_t)g3[0], (half_t)g3[1], (half_t)g3[2], (half_t)g3[3]};
            v4h h4 = {(half_t)g4[0], (half_t)g4[1], (half_t)g4[2], (half_t)g4[3]};
            *(v4h*)(s_w + i)        = h2;        // ds_store_b64
            *(v4h*)(s_w + 4096 + i) = h3;
            *(v4h*)(s_w + 8192 + i) = h4;
        }
        if (tid < 64) {
            s_w1p[tid * 4 + 0] = W1[k * 192 + tid * 3 + 0];
            s_w1p[tid * 4 + 1] = W1[k * 192 + tid * 3 + 1];
            s_w1p[tid * 4 + 2] = W1[k * 192 + tid * 3 + 2];
            s_w1p[tid * 4 + 3] = b1[k * 64 + tid];
        } else if (tid < 128) {
            int c = tid - 64;
            s_b[c]       = b2[k * 64 + c];
            s_b[64 + c]  = b3[k * 64 + c];
            s_b[128 + c] = b4[k * 64 + c];
            s_w5[c]      = W5[k * 64 + c];
        } else if (tid == 128) {
            s_misc[0] = b5[k];
            s_misc[1] = scales[k * 4 + 0];
            s_misc[2] = scales[k * 4 + 1];
            s_misc[3] = scales[k * 4 + 2];
            s_misc[4] = scales[k * 4 + 3];
        }
        __syncthreads();

        // ---- layer 1: K=3, pure VALU; lane <-> point, write f16 act rows ----
        const float sc1 = s_misc[1];
        #pragma unroll 8
        for (int c = 0; c < 64; ++c) {
            v4f wb = *(const v4f*)(s_w1p + c * 4);   // one ds_load_b128 per neuron
            float h = wb[3];
            h = fmaf(f0, wb[0], h);
            h = fmaf(f1, wb[1], h);
            h = fmaf(tv, wb[2], h);
            act[lane * 64 + c] = (half_t)fast_tanh(sc1 * h);
        }

        // ---- layers 2..4: WMMA f16 -> f32, 32x64 tile per wave ----
        #pragma unroll
        for (int L = 0; L < 3; ++L) {
            const float  sc   = s_misc[2 + L];
            const float* bias = s_b + 64 * L;

            // A fragments (16x32 f16 layout): row = mt*16+r, K-halves split by hi.
            VA a00, a01, a10, a11;
            {
                const half_t* pa = act + r * 64 + hi * 8;
                a00.h[0] = *(const v8h*)(pa);        a00.h[1] = *(const v8h*)(pa + 16);
                a01.h[0] = *(const v8h*)(pa + 32);   a01.h[1] = *(const v8h*)(pa + 48);
                const half_t* pb = pa + 16 * 64;
                a10.h[0] = *(const v8h*)(pb);        a10.h[1] = *(const v8h*)(pb + 16);
                a11.h[0] = *(const v8h*)(pb + 32);   a11.h[1] = *(const v8h*)(pb + 48);
            }

            #pragma unroll
            for (int nt = 0; nt < 4; ++nt) {
                const int col = nt * 16 + r;          // N index = lane%16
                const float sb = sc * bias[col];      // bias folded into activation FMA

                // B fragments (32x16): B[k][o] = W[o][k]; K-halves split by hi (same order as A).
                VA b0, b1v;
                const half_t* wp = s_w + L * 4096 + col * 64 + hi * 8;
                b0.h[0]  = *(const v8h*)(wp);        b0.h[1]  = *(const v8h*)(wp + 16);
                b1v.h[0] = *(const v8h*)(wp + 32);   b1v.h[1] = *(const v8h*)(wp + 48);

                v8f acc0, acc1;
                acc0 = __builtin_amdgcn_wmma_f32_16x16x32_f16(false, a00.v, false, b0.v,  (short)0, vzero, false, false);
                acc0 = __builtin_amdgcn_wmma_f32_16x16x32_f16(false, a01.v, false, b1v.v, (short)0, acc0,  false, false);
                acc1 = __builtin_amdgcn_wmma_f32_16x16x32_f16(false, a10.v, false, b0.v,  (short)0, vzero, false, false);
                acc1 = __builtin_amdgcn_wmma_f32_16x16x32_f16(false, a11.v, false, b1v.v, (short)0, acc1,  false, false);

                // ScaledTanh + writeback per documented C/D layout (VGPR i -> row).
                #pragma unroll
                for (int i = 0; i < 8; ++i) {
                    const int m0 = hi * 8 + i;
                    act[m0 * 64 + col]        = (half_t)fast_tanh(fmaf(sc, acc0[i], sb));
                    act[(16 + m0) * 64 + col] = (half_t)fast_tanh(fmaf(sc, acc1[i], sb));
                }
            }
        }

        // ---- layer 5 (64 -> 1) + gated accumulate ----
        float o = s_misc[0];
        const half_t* row = act + lane * 64;
        #pragma unroll
        for (int c = 0; c < 64; c += 8) {
            v8h hv = *(const v8h*)(row + c);
            v4f w0 = *(const v4f*)(s_w5 + c);
            v4f w1 = *(const v4f*)(s_w5 + c + 4);
            o = fmaf((float)hv[0], w0[0], o);
            o = fmaf((float)hv[1], w0[1], o);
            o = fmaf((float)hv[2], w0[2], o);
            o = fmaf((float)hv[3], w0[3], o);
            o = fmaf((float)hv[4], w1[0], o);
            o = fmaf((float)hv[5], w1[1], o);
            o = fmaf((float)hv[6], w1[2], o);
            o = fmaf((float)hv[7], w1[3], o);
        }
        {   // recompute this subnet's logit (cheap; avoids scratch array)
            float gx = __expf(lgam[2 * k]);
            float gt = __expf(lgam[2 * k + 1]);
            float dx = xv - centers[2 * k];
            float dt = tv - centers[2 * k + 1];
            float lk = -(gx * dx * dx + gt * dt * dt);
            u += __expf(lk - lmax) * invS * o;
        }
    }

    const float res = fast_tanh(tv) * u + xv * xv * f0;   // u0 = x^2 cos(pi x)
    if (p < n) out[p] = res;
}

extern "C" void kernel_launch(void* const* d_in, const int* in_sizes, int n_in,
                              void* d_out, int out_size, void* d_ws, size_t ws_size,
                              hipStream_t stream) {
    (void)n_in; (void)d_ws; (void)ws_size; (void)out_size;
    const float* x       = (const float*)d_in[0];
    const float* t       = (const float*)d_in[1];
    const float* W1      = (const float*)d_in[2];
    const float* b1      = (const float*)d_in[3];
    const float* W2      = (const float*)d_in[4];
    const float* b2      = (const float*)d_in[5];
    const float* W3      = (const float*)d_in[6];
    const float* b3      = (const float*)d_in[7];
    const float* W4      = (const float*)d_in[8];
    const float* b4      = (const float*)d_in[9];
    const float* W5      = (const float*)d_in[10];
    const float* b5      = (const float*)d_in[11];
    const float* scales  = (const float*)d_in[12];
    const float* centers = (const float*)d_in[13];
    const float* lgam    = (const float*)d_in[14];
    float* out = (float*)d_out;

    const int n = in_sizes[0];
    if (n <= 0) return;
    const int grid = (n + 255) / 256;
    pinn_kernel<<<grid, 256, 0, stream>>>(x, t, W1, b1, W2, b2, W3, b3, W4, b4,
                                          W5, b5, scales, centers, lgam, out, n);
}
